// DKVMN_81235011436941
// MI455X (gfx1250) — matrix operand order repaired and checked by reading, not compiled, not run on hardware
//
#include <hip/hip_runtime.h>
#include <hip/hip_bf16.h>
#include <math.h>

#define BQ 20000      // Qn
#define BB 128        // batch
#define SS 200        // seq len
#define VV 128        // value dim
#define KK 128        // key dim
#define CC 64         // concepts
#define SUMN 128      // summary dim
#define NPOS (BB * SS) // 25600 flat positions

typedef __attribute__((ext_vector_type(16))) _Float16 v16h;
typedef __attribute__((ext_vector_type(8)))  float    v8f;
typedef __attribute__((ext_vector_type(4)))  float    v4f;

__device__ __forceinline__ v8f wmma_f32_f16(v16h a, v16h b, v8f c) {
  return __builtin_amdgcn_wmma_f32_16x16x32_f16(
      false, a, false, b, (short)0, c, false, false);
}

// Fast activations: v_exp_f32 + v_rcp_f32, correct saturation at +-inf.
__device__ __forceinline__ float fast_sigmoid(float x) {
  return __builtin_amdgcn_rcpf(1.0f + __expf(-x));
}
__device__ __forceinline__ float fast_tanh(float x) {
  return 1.0f - 2.0f * __builtin_amdgcn_rcpf(__expf(2.0f * x) + 1.0f);
}

// A fragment (16x32 f16 tile) from an f16 row pointer. Lane L holds row
// m = L%16. Element i -> K per ISA 16-bit A layout:
//   j=i/2, h=i%2; K = (j<4 ? 2j+h : 16+2(j-4)+h) + (L>=16 ? 8 : 0)
__device__ __forceinline__ v16h load_a_frag_h(const _Float16* rp, int k0, int lane) {
  int kadd = (lane >= 16) ? 8 : 0;
  v16h a;
#pragma unroll
  for (int i = 0; i < 16; ++i) {
    int j = i >> 1, h = i & 1;
    int k = (j < 4) ? (2 * j + h) : (16 + 2 * (j - 4) + h);
    a[i] = rp[k0 + k + kadd];
  }
  return a;
}

// Same, from an f32 row (only k_summ's READ rows).
__device__ __forceinline__ v16h load_a_frag_f(const float* rp, int k0, int lane) {
  int kadd = (lane >= 16) ? 8 : 0;
  v16h a;
#pragma unroll
  for (int i = 0; i < 16; ++i) {
    int j = i >> 1, h = i & 1;
    int k = (j < 4) ? (2 * j + h) : (16 + 2 * (j - 4) + h);
    a[i] = (_Float16)rp[k0 + k + kadd];
  }
  return a;
}

// B fragment built from a row-major KxN f32 matrix (prep-time only).
__device__ __forceinline__ v16h build_b_frag(const float* W, int ldn, int k0,
                                             int n0, int lane) {
  const float* p = W + (size_t)(k0 + ((lane >= 16) ? 8 : 0)) * ldn
                     + n0 + (lane & 15);
  v16h b;
#pragma unroll
  for (int i = 0; i < 16; ++i) b[i] = (_Float16)p[(size_t)(i < 8 ? i : i + 8) * ldn];
  return b;
}

// ---------------------------------------------------------------------------
// Prep kernels
// ---------------------------------------------------------------------------
__global__ void k_cvt_f32_f16(const float* __restrict__ src,
                              _Float16* __restrict__ dst, int n) {
  int i = blockIdx.x * blockDim.x + threadIdx.x;
  if (i < n) dst[i] = (_Float16)src[i];
}

__global__ void __launch_bounds__(32)
k_pack_b(const float* __restrict__ W, int ldn, int NT, v16h* __restrict__ out) {
  int lane = threadIdx.x;
  int kt = blockIdx.x / NT;
  int nt = blockIdx.x % NT;
  out[(size_t)blockIdx.x * 32 + lane] = build_b_frag(W, ldn, kt * 32, nt * 16, lane);
}

// ---------------------------------------------------------------------------
// Kernel 1: per-position precompute. One wave per TWO 16-row tiles.
// Outputs are stored COLUMN-MAJOR (Et/At: [n][row], Wt: [c][row]) so each
// lane's 8 consecutive D-rows merge into two aligned b128 stores.
// ---------------------------------------------------------------------------
__global__ void __launch_bounds__(32)
k_precompute(const _Float16* __restrict__ IEH, const _Float16* __restrict__ QEH,
             const v16h* __restrict__ PBe, const v16h* __restrict__ PBa,
             const v16h* __restrict__ PBk,
             const float* __restrict__ erase_b, const float* __restrict__ add_b,
             const int* __restrict__ input,
             float* __restrict__ Et, float* __restrict__ At,
             float* __restrict__ Wt) {
  const int lane  = threadIdx.x;
  const int m     = lane & 15;
  const int outm  = (lane >= 16) ? 8 : 0;
  const int tbase = blockIdx.x * 2;

  const _Float16* ivp[2];
  const _Float16* qp[2];
#pragma unroll
  for (int t = 0; t < 2; ++t) {
    int row  = (tbase + t) * 16 + m;
    int idx  = input[row];
    int qidx = (idx > BQ) ? idx - BQ : idx;     // qid = input - Qn*(input>Qn)
    ivp[t] = IEH + (size_t)idx  * VV;
    qp[t]  = QEH + (size_t)qidx * KK;
  }

  // -------- phase 1: e = sigmoid(iv@We+b), a = tanh(iv@Wa+b) --------
  v16h ai[2][4];
#pragma unroll
  for (int t = 0; t < 2; ++t)
#pragma unroll
    for (int kt = 0; kt < 4; ++kt)
      ai[t][kt] = load_a_frag_h(ivp[t], kt * 32, lane);

#pragma unroll
  for (int nt = 0; nt < 8; ++nt) {
    v8f ce[2] = {{}, {}};
    v8f ca[2] = {{}, {}};
#pragma unroll
    for (int kt = 0; kt < 4; ++kt) {
      v16h be = PBe[(size_t)(kt * 8 + nt) * 32 + lane];   // 32B vector load
      v16h ba = PBa[(size_t)(kt * 8 + nt) * 32 + lane];
#pragma unroll
      for (int t = 0; t < 2; ++t) {
        ce[t] = wmma_f32_f16(ai[t][kt], be, ce[t]);
        ca[t] = wmma_f32_f16(ai[t][kt], ba, ca[t]);
      }
    }
    int n = nt * 16 + m;
    float eb = erase_b[n], ab = add_b[n];
#pragma unroll
    for (int t = 0; t < 2; ++t) {
      v4f e0, e1, a0, a1;
#pragma unroll
      for (int r = 0; r < 4; ++r) {
        e0[r] = fast_sigmoid(ce[t][r] + eb);
        e1[r] = fast_sigmoid(ce[t][r + 4] + eb);
        a0[r] = fast_tanh(ca[t][r] + ab);
        a1[r] = fast_tanh(ca[t][r + 4] + ab);
      }
      size_t base = (size_t)n * NPOS + (tbase + t) * 16 + outm;
      *(v4f*)(Et + base)     = e0;
      *(v4f*)(Et + base + 4) = e1;
      *(v4f*)(At + base)     = a0;
      *(v4f*)(At + base + 4) = a1;
    }
  }

  // -------- phase 2: attention logits q @ key_memory --------
  v16h aq[2][4];
#pragma unroll
  for (int t = 0; t < 2; ++t)
#pragma unroll
    for (int kt = 0; kt < 4; ++kt)
      aq[t][kt] = load_a_frag_h(qp[t], kt * 32, lane);

#pragma unroll
  for (int nt = 0; nt < 4; ++nt) {
    v8f c[2] = {{}, {}};
#pragma unroll
    for (int kt = 0; kt < 4; ++kt) {
      v16h b = PBk[(size_t)(kt * 4 + nt) * 32 + lane];
#pragma unroll
      for (int t = 0; t < 2; ++t) c[t] = wmma_f32_f16(aq[t][kt], b, c[t]);
    }
    int n = nt * 16 + m;
#pragma unroll
    for (int t = 0; t < 2; ++t) {
      v4f c0, c1;
#pragma unroll
      for (int r = 0; r < 4; ++r) { c0[r] = c[t][r]; c1[r] = c[t][r + 4]; }
      size_t base = (size_t)n * NPOS + (tbase + t) * 16 + outm;
      *(v4f*)(Wt + base)     = c0;
      *(v4f*)(Wt + base + 4) = c1;
    }
  }
}

// ---------------------------------------------------------------------------
// Kernel 2: generic strided row softmax (in place).
// Element (r, c) lives at X[r*row_stride + c*col_stride].
// ---------------------------------------------------------------------------
__global__ void k_softmax_s(float* __restrict__ X, int nrows, int ncols,
                            long row_stride, long col_stride) {
  int r = blockIdx.x * blockDim.x + threadIdx.x;
  if (r >= nrows) return;
  float* p = X + (size_t)r * row_stride;
  float mx = -1e30f;
  for (int c = 0; c < ncols; ++c) mx = fmaxf(mx, p[(size_t)c * col_stride]);
  float s = 0.0f;
  for (int c = 0; c < ncols; ++c) {
    float v = __expf(p[(size_t)c * col_stride] - mx);
    p[(size_t)c * col_stride] = v;
    s += v;
  }
  float inv = __builtin_amdgcn_rcpf(s);
  for (int c = 0; c < ncols; ++c) p[(size_t)c * col_stride] *= inv;
}

// ---------------------------------------------------------------------------
// Kernel 3: target attention logits wt_log = q_emb[target_id] @ key_memory.
// WT kept row-major (tiny; consumed row-wise by k_scan).
// ---------------------------------------------------------------------------
__global__ void __launch_bounds__(32)
k_wt_logits(const _Float16* __restrict__ QEH, const v16h* __restrict__ PBk,
            const int* __restrict__ target_id, float* __restrict__ WT) {
  const int lane = threadIdx.x;
  const int tile = blockIdx.x;
  const int m    = lane & 15;
  const int row  = tile * 16 + m;                       // b
  const _Float16* qp = QEH + (size_t)target_id[row] * KK;
  v16h aq[4];
#pragma unroll
  for (int kt = 0; kt < 4; ++kt) aq[kt] = load_a_frag_h(qp, kt * 32, lane);
  const int outm = (lane >= 16) ? 8 : 0;
#pragma unroll
  for (int nt = 0; nt < 4; ++nt) {
    v8f c = {};
#pragma unroll
    for (int kt = 0; kt < 4; ++kt)
      c = wmma_f32_f16(aq[kt], PBk[(size_t)(kt * 4 + nt) * 32 + lane], c);
    int n = nt * 16 + m;
#pragma unroll
    for (int r = 0; r < 8; ++r)
      WT[(size_t)(tile * 16 + outm + r) * CC + n] = c[r];
  }
}

// ---------------------------------------------------------------------------
// Kernel 4: sequential memory scan + final read. One block per batch element;
// mem (128x64) in registers. Transposed operands: each thread streams its own
// contiguous 200-float row (1 cacheline per 16 steps); w staged via LDS.
// ---------------------------------------------------------------------------
__global__ void __launch_bounds__(256)
k_scan(const float* __restrict__ init_mem,
       const float* __restrict__ Et, const float* __restrict__ At,
       const float* __restrict__ Wt, const float* __restrict__ WT,
       float* __restrict__ READ) {
  const int b   = blockIdx.x;
  const int tid = threadIdx.x;
  const int v   = tid & 127;      // value row owned by this thread
  const int ch  = tid >> 7;       // which half of the 64 concepts
  float mreg[32];
#pragma unroll
  for (int i = 0; i < 32; ++i) mreg[i] = init_mem[v * CC + ch * 32 + i];

  __shared__ float sw[CC];
  __shared__ float red[256];
  const size_t rowbase = (size_t)b * SS;
  const float* ep = Et + (size_t)v * NPOS + rowbase;
  const float* ap = At + (size_t)v * NPOS + rowbase;

  for (int s = 0; s < SS; ++s) {
    if (tid < CC) sw[tid] = Wt[(size_t)tid * NPOS + rowbase + s];
    __syncthreads();
    float ev = ep[s];
    float av = ap[s];
    __builtin_prefetch(ep + s + 16, 0, 1);              // next cacheline
    __builtin_prefetch(ap + s + 16, 0, 1);
#pragma unroll
    for (int i = 0; i < 32; ++i) {
      float wv = sw[ch * 32 + i];
      mreg[i] = mreg[i] * (1.0f - ev * wv) + av * wv;
    }
    __syncthreads();
  }

  float part = 0.0f;
  const float* wt = WT + (size_t)b * CC + ch * 32;
#pragma unroll
  for (int i = 0; i < 32; ++i) part += mreg[i] * wt[i];
  red[tid] = part;
  __syncthreads();
  if (tid < VV) READ[(size_t)b * VV + tid] = red[tid] + red[tid + 128];
}

// ---------------------------------------------------------------------------
// Kernel 5: summ = tanh([read, qv] @ summ_W + summ_b) via WMMA (K=256).
// ---------------------------------------------------------------------------
__global__ void __launch_bounds__(32)
k_summ(const float* __restrict__ READ, const _Float16* __restrict__ QEH,
       const int* __restrict__ target_id,
       const v16h* __restrict__ PBs, const float* __restrict__ summ_b,
       float* __restrict__ SUMMo) {
  const int lane = threadIdx.x;
  const int tile = blockIdx.x;
  const int m    = lane & 15;
  const int row  = tile * 16 + m;                       // b
  const float*    rp = READ + (size_t)row * VV;
  const _Float16* qp = QEH + (size_t)target_id[row] * KK;
  v16h af[8];
#pragma unroll
  for (int kt = 0; kt < 4; ++kt) {
    af[kt]     = load_a_frag_f(rp, kt * 32, lane);      // k in [0,128): read
    af[kt + 4] = load_a_frag_h(qp, kt * 32, lane);      // k in [128,256): qv
  }
  const int outm = (lane >= 16) ? 8 : 0;
#pragma unroll
  for (int nt = 0; nt < 8; ++nt) {
    v8f c = {};
#pragma unroll
    for (int kt = 0; kt < 8; ++kt)
      c = wmma_f32_f16(af[kt], PBs[(size_t)(kt * 8 + nt) * 32 + lane], c);
    int n = nt * 16 + m;
    float sb = summ_b[n];
#pragma unroll
    for (int r = 0; r < 8; ++r)
      SUMMo[(size_t)(tile * 16 + outm + r) * SUMN + n] = fast_tanh(c[r] + sb);
  }
}

// ---------------------------------------------------------------------------
// Kernel 6: out = summ @ out_W + out_b   (B x 1)
// ---------------------------------------------------------------------------
__global__ void __launch_bounds__(128)
k_out(const float* __restrict__ SUMMo, const float* __restrict__ out_W,
      const float* __restrict__ out_b, float* __restrict__ out) {
  int b = threadIdx.x;
  float s = 0.0f;
#pragma unroll 8
  for (int j = 0; j < SUMN; ++j) s += SUMMo[(size_t)b * SUMN + j] * out_W[j];
  out[b] = s + out_b[0];
}

extern "C" void kernel_launch(void* const* d_in, const int* in_sizes, int n_in,
                              void* d_out, int out_size, void* d_ws, size_t ws_size,
                              hipStream_t stream) {
  const float* q_emb      = (const float*)d_in[0];   // (20001,128)
  const float* i_emb      = (const float*)d_in[1];   // (40001,128)
  const float* key_memory = (const float*)d_in[2];   // (128,64)
  const float* init_mem   = (const float*)d_in[3];   // (128,64)
  const float* erase_W    = (const float*)d_in[4];   // (128,128)
  const float* erase_b    = (const float*)d_in[5];   // (128)
  const float* add_W      = (const float*)d_in[6];   // (128,128)
  const float* add_b      = (const float*)d_in[7];   // (128)
  const float* summ_W     = (const float*)d_in[8];   // (256,128)
  const float* summ_b     = (const float*)d_in[9];   // (128)
  const float* out_W      = (const float*)d_in[10];  // (128,1)
  const float* out_b      = (const float*)d_in[11];  // (1)
  const int*   input      = (const int*)d_in[12];    // (128,200)
  const int*   target_id  = (const int*)d_in[13];    // (128)

  const size_t N  = NPOS;                            // 25600 positions
  const int    ni = in_sizes[1];                     // 40001*128
  const int    nq = in_sizes[0];                     // 20001*128

  uintptr_t p = ((uintptr_t)d_ws + 255) & ~(uintptr_t)255;
  auto carve = [&p](size_t bytes) -> void* {
    void* r = (void*)p;
    p += (bytes + 255) & ~(size_t)255;
    return r;
  };
  float*     Et    = (float*)carve(N * VV * 4);      // [n][row] column-major
  float*     At    = (float*)carve(N * VV * 4);
  float*     Wt    = (float*)carve(N * CC * 4);      // [c][row]
  float*     WT    = (float*)carve((size_t)BB * CC * 4);
  float*     READ  = (float*)carve((size_t)BB * VV * 4);
  float*     SUMMo = (float*)carve((size_t)BB * SUMN * 4);
  _Float16*  IEH   = (_Float16*)carve((size_t)ni * 2);
  _Float16*  QEH   = (_Float16*)carve((size_t)nq * 2);
  v16h*      PBe   = (v16h*)carve(32 * 32 * sizeof(v16h));   // 4kt x 8nt
  v16h*      PBa   = (v16h*)carve(32 * 32 * sizeof(v16h));
  v16h*      PBk   = (v16h*)carve(16 * 32 * sizeof(v16h));   // 4kt x 4nt
  v16h*      PBs   = (v16h*)carve(64 * 32 * sizeof(v16h));   // 8kt x 8nt

  // Prep: f16 tables + packed B fragments (L2-resident afterwards)
  k_cvt_f32_f16<<<(ni + 255) / 256, 256, 0, stream>>>(i_emb, IEH, ni);
  k_cvt_f32_f16<<<(nq + 255) / 256, 256, 0, stream>>>(q_emb, QEH, nq);
  k_pack_b<<<32, 32, 0, stream>>>(erase_W, VV, 8, PBe);
  k_pack_b<<<32, 32, 0, stream>>>(add_W, VV, 8, PBa);
  k_pack_b<<<16, 32, 0, stream>>>(key_memory, CC, 4, PBk);
  k_pack_b<<<64, 32, 0, stream>>>(summ_W, SUMN, 8, PBs);

  // Main pipeline
  k_precompute<<<(int)(N / 32), 32, 0, stream>>>(
      IEH, QEH, PBe, PBa, PBk, erase_b, add_b, input, Et, At, Wt);
  k_softmax_s<<<(int)((N + 255) / 256), 256, 0, stream>>>(
      Wt, (int)N, CC, 1L, (long)N);                  // transposed: coalesced
  k_wt_logits<<<BB / 16, 32, 0, stream>>>(QEH, PBk, target_id, WT);
  k_softmax_s<<<1, 128, 0, stream>>>(WT, BB, CC, (long)CC, 1L);
  k_scan<<<BB, 256, 0, stream>>>(init_mem, Et, At, Wt, WT, READ);
  k_summ<<<BB / 16, 32, 0, stream>>>(READ, QEH, target_id, PBs, summ_b, SUMMo);
  k_out<<<1, 128, 0, stream>>>(SUMMo, out_W, out_b, (float*)d_out);
}
